// TemporalMambaBlock_2731599200643
// MI455X (gfx1250) — matrix operand order repaired and checked by reading, not compiled, not run on hardware
//
#include <hip/hip_runtime.h>

// ---------------- problem constants ----------------
#define NB 4
#define TT 2048
#define DM 512
#define DI 1024
#define DSTATE 16
#define DTR 32
#define BT (NB * TT)

typedef __bf16 bf16;
typedef __attribute__((ext_vector_type(16))) __bf16 v16bf;
typedef __attribute__((ext_vector_type(8)))  __bf16 v8bf;
typedef __attribute__((ext_vector_type(8)))  float  v8f;

__device__ __forceinline__ float bf2f(bf16 v) { return (float)v; }
__device__ __forceinline__ bf16  f2bf(float f) { return (bf16)f; }

union bfrag { v16bf v; v8bf h[2]; };

// Load one 16x32 A fragment + four 32x16 B fragments for K-step `kk`.
#define LOADF(A_, B_, kk)                                                   \
  do {                                                                      \
    (A_).h[0] = *(const v8bf*)(ap + (kk) + sel * 8);                        \
    (A_).h[1] = *(const v8bf*)(ap + (kk) + 16 + sel * 8);                   \
    _Pragma("unroll") for (int j_ = 0; j_ < 4; ++j_) {                      \
      const bf16* bj_ = bp + (size_t)j_ * 16 * ldb + (kk) + sel * 8;        \
      (B_)[j_].h[0] = *(const v8bf*)(bj_);                                  \
      (B_)[j_].h[1] = *(const v8bf*)(bj_ + 16);                             \
    }                                                                       \
  } while (0)

// Issue the 4 WMMAs of one K-step (shared A fragment).
#define WMMAF(A_, B_)                                                       \
  do {                                                                      \
    _Pragma("unroll") for (int j_ = 0; j_ < 4; ++j_)                        \
      acc[j_] = __builtin_amdgcn_wmma_f32_16x16x32_bf16(                    \
          false, (A_).v, false, (B_)[j_].v, (short)0, acc[j_], false, false); \
  } while (0)

#if __has_builtin(__builtin_amdgcn_sched_group_barrier)
// stage = 10 VMEM_READ (0x20) then 4 MFMA/WMMA (0x8)
#define SCHED_STAGE()                                    \
  do {                                                   \
    __builtin_amdgcn_sched_group_barrier(0x020, 10, 0);  \
    __builtin_amdgcn_sched_group_barrier(0x008, 4, 0);   \
    __builtin_amdgcn_sched_group_barrier(0x020, 10, 0);  \
    __builtin_amdgcn_sched_group_barrier(0x008, 4, 0);   \
  } while (0)
#else
#define SCHED_STAGE() do { } while (0)
#endif

// ---------------- WMMA 16x64 strip GEMM core ----------------
// A: M x K row-major bf16 (lda), Bt: N x K row-major bf16 (ldb) = B transposed.
// One wave computes a 16(M) x 64(N) strip with explicit double-buffering:
// fragments for K-step k+32 are loaded while WMMAs for step k execute.
__device__ __forceinline__ void wmma_strip4(const bf16* __restrict__ A, int lda,
                                            const bf16* __restrict__ Bt, int ldb,
                                            int K, int m0, int n0, v8f acc[4]) {
  const int lane = threadIdx.x & 31;
  const int idx  = lane & 15;
  const int sel  = lane >> 4;
  const bf16* ap = A  + (size_t)(m0 + idx) * lda;
  const bf16* bp = Bt + (size_t)(n0 + idx) * ldb;
#pragma unroll
  for (int j = 0; j < 4; ++j)
    acc[j] = (v8f){0.f, 0.f, 0.f, 0.f, 0.f, 0.f, 0.f, 0.f};

  bfrag a0, a1;
  bfrag b0[4], b1[4];
  LOADF(a0, b0, 0);
  int k = 0;
  // main pipeline: branch-free body, two K-steps per iteration
  for (; k + 128 <= K; k += 64) {
    LOADF(a1, b1, k + 32);
    WMMAF(a0, b0);
    LOADF(a0, b0, k + 64);
    WMMAF(a1, b1);
    SCHED_STAGE();
  }
  if (k + 64 <= K) {         // 64 columns of K remain
    LOADF(a1, b1, k + 32);
    WMMAF(a0, b0);
    WMMAF(a1, b1);
  } else {                   // 32 columns of K remain (e.g. K=32)
    WMMAF(a0, b0);
  }
}

// wave -> 16x64 strip coordinates; waves-per-block derived from blockDim.
#define STRIP_COORDS()                                     \
  const int wid  = threadIdx.x >> 5;                       \
  const int wpb  = blockDim.x >> 5;                        \
  const int m0   = blockIdx.x * 16;                        \
  const int n0   = (blockIdx.y * wpb + wid) * 64;          \
  const int lane = threadIdx.x & 31;                       \
  const int idx  = lane & 15;                              \
  const int sel  = lane >> 4;

// ---------------- kernels ----------------

// RMSNorm -> bf16, also writes the time-reversed copy for the bwd branch.
__global__ void k_rmsnorm(const float* __restrict__ x, const float* __restrict__ w,
                          bf16* __restrict__ xn, bf16* __restrict__ xnr) {
  const int row = blockIdx.x;               // b*TT + t
  const int b = row / TT, t = row % TT;
  const float* xr = x + (size_t)row * DM;
  __shared__ float red[8];
  __shared__ float rr_s;
  float s = 0.f;
  for (int i = threadIdx.x; i < DM; i += blockDim.x) { float v = xr[i]; s += v * v; }
  for (int o = 16; o > 0; o >>= 1) s += __shfl_down(s, o, 32);
  if ((threadIdx.x & 31) == 0) red[threadIdx.x >> 5] = s;
  __syncthreads();
  if (threadIdx.x == 0) {
    float tot = 0.f;
    for (int i = 0; i < (int)(blockDim.x >> 5); ++i) tot += red[i];
    rr_s = rsqrtf(tot / (float)DM + 1e-6f);
  }
  __syncthreads();
  const float rr = rr_s;
  const size_t rrow = (size_t)(b * TT + (TT - 1 - t)) * DM;
  for (int i = threadIdx.x; i < DM; i += blockDim.x) {
    bf16 v = f2bf(xr[i] * rr * w[i]);
    xn[(size_t)row * DM + i] = v;
    xnr[rrow + i] = v;
  }
}

// fp32 (K x N) weight -> bf16 (N x K) transposed copy.
__global__ void k_cvt_t(const float* __restrict__ w, bf16* __restrict__ wt,
                        int K, int N) {
  int i = blockIdx.x * blockDim.x + threadIdx.x;
  if (i >= K * N) return;
  int k = i / N, n = i - k * N;
  wt[(size_t)n * K + k] = f2bf(w[i]);
}

// xz = xn @ w_in  (M=BT, N=2*DI, K=DM); split columns into xc / z.
// The xc/z boundary (DI) is strip-tile aligned -> uniform (scalar) branch.
__global__ void k_gemm_xz(const bf16* __restrict__ A, const bf16* __restrict__ Bt,
                          bf16* __restrict__ xc, bf16* __restrict__ z) {
  STRIP_COORDS();
  v8f acc[4];
  wmma_strip4(A, DM, Bt, DM, DM, m0, n0, acc);
#pragma unroll
  for (int j = 0; j < 4; ++j) {
    const int tcol = n0 + j * 16;                 // uniform per tile
    bf16* dst = (tcol < DI) ? xc : z;
    const int gcol = (tcol < DI ? tcol : tcol - DI) + idx;
#pragma unroll
    for (int r = 0; r < 8; ++r) {
      const int grow = m0 + sel * 8 + r;
      dst[(size_t)grow * DI + gcol] = f2bf(acc[j][r]);
    }
  }
}

// depthwise causal conv (D_CONV=4) + bias + SiLU
__global__ void k_conv(const bf16* __restrict__ xc, const float* __restrict__ cw,
                       const float* __restrict__ cb, bf16* __restrict__ xconv) {
  int i = blockIdx.x * blockDim.x + threadIdx.x;
  if (i >= BT * DI) return;
  const int d = i & (DI - 1);
  const int row = i >> 10;
  const int t = row & (TT - 1);
  const int b = row >> 11;
  float acc = cb[d];
#pragma unroll
  for (int k = 0; k < 4; ++k) {
    int tk = t - 3 + k;
    if (tk >= 0) acc += cw[d * 4 + k] * bf2f(xc[(size_t)(b * TT + tk) * DI + d]);
  }
  float sg = 1.f / (1.f + __expf(-acc));
  xconv[i] = f2bf(acc * sg);
}

// proj = xconv @ w_xproj (M=BT, N=64, K=DI); also emit bf16 dt_in (cols<32).
__global__ void k_gemm_proj(const bf16* __restrict__ A, const bf16* __restrict__ Bt,
                            float* __restrict__ proj, bf16* __restrict__ dtin) {
  STRIP_COORDS();
  v8f acc[4];
  wmma_strip4(A, DI, Bt, DI, DI, m0, n0, acc);
#pragma unroll
  for (int j = 0; j < 4; ++j) {
    const int tcol = n0 + j * 16;                 // uniform per tile
    const int gcol = tcol + idx;
#pragma unroll
    for (int r = 0; r < 8; ++r) {
      const int grow = m0 + sel * 8 + r;
      float v = acc[j][r];
      proj[(size_t)grow * 64 + gcol] = v;
      if (tcol < DTR) dtin[(size_t)grow * DTR + gcol] = f2bf(v);
    }
  }
}

// dt = softplus(dt_in @ w_dt + b_dt)  (M=BT, N=DI, K=32 -> single WMMA step/tile)
__global__ void k_gemm_dt(const bf16* __restrict__ A, const bf16* __restrict__ Bt,
                          const float* __restrict__ bdt, float* __restrict__ dt) {
  STRIP_COORDS();
  v8f acc[4];
  wmma_strip4(A, DTR, Bt, DTR, DTR, m0, n0, acc);
#pragma unroll
  for (int j = 0; j < 4; ++j) {
    const int gcol = n0 + j * 16 + idx;
    const float bias = bdt[gcol];
#pragma unroll
    for (int r = 0; r < 8; ++r) {
      const int grow = m0 + sel * 8 + r;
      float v = acc[j][r] + bias;
      float sp = (v > 20.f) ? v : log1pf(__expf(v));
      dt[(size_t)grow * DI + gcol] = sp;
    }
  }
}

// selective scan + fused gating; for dir==1 writes time-unflipped rows.
__global__ void k_scan(const float* __restrict__ proj, const float* __restrict__ dtb,
                       const bf16* __restrict__ xconv, const bf16* __restrict__ zb,
                       const float* __restrict__ A_log, const float* __restrict__ Dv,
                       bf16* __restrict__ yg, int dir) {
  const int b = blockIdx.x;
  const int d = blockIdx.y * blockDim.x + threadIdx.x;
  float Ac[DSTATE], h[DSTATE];
#pragma unroll
  for (int s = 0; s < DSTATE; ++s) {
    Ac[s] = -__expf(A_log[d * DSTATE + s]);
    h[s] = 0.f;
  }
  const float Dd = Dv[d];
  __shared__ float sB[DSTATE], sC[DSTATE];
  for (int t = 0; t < TT; ++t) {
    const int row = b * TT + t;
    __syncthreads();
    if (threadIdx.x < 32) {
      float v = proj[(size_t)row * 64 + DTR + threadIdx.x];
      if (threadIdx.x < DSTATE) sB[threadIdx.x] = v;
      else                      sC[threadIdx.x - DSTATE] = v;
    }
    __syncthreads();
    const float dt  = dtb[(size_t)row * DI + d];
    const float xv  = bf2f(xconv[(size_t)row * DI + d]);
    const float dtx = dt * xv;
    float y = 0.f;
#pragma unroll
    for (int s = 0; s < DSTATE; ++s) {
      h[s] = __expf(dt * Ac[s]) * h[s] + dtx * sB[s];
      y += h[s] * sC[s];
    }
    const float zv = bf2f(zb[(size_t)row * DI + d]);
    const float sg = 1.f / (1.f + __expf(-zv));
    const float ygv = (y + Dd * xv) * (zv * sg);
    const int orow = dir ? (b * TT + (TT - 1 - t)) : row;
    yg[(size_t)orow * DI + d] = f2bf(ygv);
  }
}

// of/ob = yg @ w_out  (M=BT, N=DM, K=DI) -> concat[:, colofs + n]
__global__ void k_gemm_out(const bf16* __restrict__ A, const bf16* __restrict__ Bt,
                           bf16* __restrict__ cat, int colofs) {
  STRIP_COORDS();
  v8f acc[4];
  wmma_strip4(A, DI, Bt, DI, DI, m0, n0, acc);
#pragma unroll
  for (int j = 0; j < 4; ++j) {
    const int gcol = colofs + n0 + j * 16 + idx;
#pragma unroll
    for (int r = 0; r < 8; ++r) {
      const int grow = m0 + sel * 8 + r;
      cat[(size_t)grow * (2 * DM) + gcol] = f2bf(acc[j][r]);
    }
  }
}

// out = x + concat @ merge_w  (M=BT, N=DM, K=2*DM)
__global__ void k_gemm_merge(const bf16* __restrict__ A, const bf16* __restrict__ Bt,
                             const float* __restrict__ x, float* __restrict__ out) {
  STRIP_COORDS();
  v8f acc[4];
  wmma_strip4(A, 2 * DM, Bt, 2 * DM, 2 * DM, m0, n0, acc);
#pragma unroll
  for (int j = 0; j < 4; ++j) {
    const int gcol = n0 + j * 16 + idx;
#pragma unroll
    for (int r = 0; r < 8; ++r) {
      const int grow = m0 + sel * 8 + r;
      out[(size_t)grow * DM + gcol] = x[(size_t)grow * DM + gcol] + acc[j][r];
    }
  }
}

// ---------------- launch ----------------
extern "C" void kernel_launch(void* const* d_in, const int* in_sizes, int n_in,
                              void* d_out, int out_size, void* d_ws, size_t ws_size,
                              hipStream_t stream) {
  (void)in_sizes; (void)n_in; (void)out_size;
  const float* x       = (const float*)d_in[0];
  const float* rms_w   = (const float*)d_in[1];
  const float* merge_w = (const float*)d_in[20];

  char* ws = (char*)d_ws;
  size_t off = 0;
  auto alloc = [&](size_t bytes) -> void* {
    void* p = ws + off;
    off = (off + bytes + 255) & ~(size_t)255;
    return p;
  };
  bf16*  xn      = (bf16*)alloc((size_t)BT * DM * 2);
  bf16*  xnr     = (bf16*)alloc((size_t)BT * DM * 2);
  bf16*  cat     = (bf16*)alloc((size_t)BT * 2 * DM * 2);
  bf16*  merge_t = (bf16*)alloc((size_t)DM * 2 * DM * 2);
  bf16*  w_in_t  = (bf16*)alloc((size_t)(2 * DI) * DM * 2);
  bf16*  w_xp_t  = (bf16*)alloc((size_t)64 * DI * 2);
  bf16*  w_dt_t  = (bf16*)alloc((size_t)DI * DTR * 2);
  bf16*  w_out_t = (bf16*)alloc((size_t)DM * DI * 2);
  bf16*  xc      = (bf16*)alloc((size_t)BT * DI * 2);
  bf16*  zb      = (bf16*)alloc((size_t)BT * DI * 2);
  bf16*  xcv     = (bf16*)alloc((size_t)BT * DI * 2);
  float* proj    = (float*)alloc((size_t)BT * 64 * 4);
  bf16*  dtin    = (bf16*)alloc((size_t)BT * DTR * 2);
  float* dtb     = (float*)alloc((size_t)BT * DI * 4);
  bf16*  yg      = (bf16*)alloc((size_t)BT * DI * 2);
  if (off > ws_size) return;  // scratch too small; bail deterministically

  k_rmsnorm<<<BT, 128, 0, stream>>>(x, rms_w, xn, xnr);
  k_cvt_t<<<((2 * DM) * DM + 255) / 256, 256, 0, stream>>>(merge_w, merge_t, 2 * DM, DM);

  for (int dir = 0; dir < 2; ++dir) {
    const int base = 2 + dir * 9;
    const float* w_in    = (const float*)d_in[base + 0];
    const float* conv_w  = (const float*)d_in[base + 1];
    const float* conv_b  = (const float*)d_in[base + 2];
    const float* w_xproj = (const float*)d_in[base + 3];
    const float* w_dt    = (const float*)d_in[base + 4];
    const float* b_dt    = (const float*)d_in[base + 5];
    const float* A_log   = (const float*)d_in[base + 6];
    const float* Dv      = (const float*)d_in[base + 7];
    const float* w_out   = (const float*)d_in[base + 8];

    k_cvt_t<<<(DM * 2 * DI + 255) / 256, 256, 0, stream>>>(w_in, w_in_t, DM, 2 * DI);
    k_cvt_t<<<(DI * 64 + 255) / 256, 256, 0, stream>>>(w_xproj, w_xp_t, DI, 64);
    k_cvt_t<<<(DTR * DI + 255) / 256, 256, 0, stream>>>(w_dt, w_dt_t, DTR, DI);
    k_cvt_t<<<(DI * DM + 255) / 256, 256, 0, stream>>>(w_out, w_out_t, DI, DM);

    // strips of 64 cols per wave; 4 waves (256 cols) per block where N allows
    k_gemm_xz<<<dim3(BT / 16, (2 * DI) / 256), 128, 0, stream>>>(
        dir ? xnr : xn, w_in_t, xc, zb);
    k_conv<<<(BT * DI + 255) / 256, 256, 0, stream>>>(xc, conv_w, conv_b, xcv);
    k_gemm_proj<<<dim3(BT / 16, 1), 32, 0, stream>>>(xcv, w_xp_t, proj, dtin);
    k_gemm_dt<<<dim3(BT / 16, DI / 256), 128, 0, stream>>>(dtin, w_dt_t, b_dt, dtb);
    k_scan<<<dim3(NB, DI / 256), 256, 0, stream>>>(proj, dtb, xcv, zb, A_log, Dv, yg, dir);
    k_gemm_out<<<dim3(BT / 16, DM / 256), 128, 0, stream>>>(yg, w_out_t, cat, dir * DM);
  }
  k_gemm_merge<<<dim3(BT / 16, DM / 256), 128, 0, stream>>>(
      cat, merge_t, x, (float*)d_out);
}
